// GRU_89842125898040
// MI455X (gfx1250) — matrix operand order repaired and checked by reading, not compiled, not run on hardware
//
#include <hip/hip_runtime.h>
#include <hip/hip_bf16.h>
#include <stdint.h>

// ---------------------------------------------------------------------------
// 3-layer GRU (batch_first), B=32 T=512 H=1024, bf16 WMMA compute / f32 state.
// Input projection: async-LDS double-buffered WMMA GEMM.
// Recurrent scan: per-timestep fused WMMA + gate kernel (weights from L2).
// ---------------------------------------------------------------------------

typedef __attribute__((ext_vector_type(16))) __bf16 v16bf;
typedef __attribute__((ext_vector_type(8)))  __bf16 v8bf;
typedef __attribute__((ext_vector_type(8)))  float  v8f;

#define B_  32
#define T_  512
#define H_  1024
#define L_  3
#define G3H 3072   // 3*H

#define LDS_STRIDE 40          // 32 K-elems padded to 40 (80B) -> bank-conflict-free
#define TILE_ROWS  192         // 64 A rows + 128 B rows per buffer

// ---------------- elementwise converts / init ------------------------------

__global__ void cvt_f32_bf16(const float* __restrict__ src,
                             __bf16* __restrict__ dst, int n) {
  int i = blockIdx.x * blockDim.x + threadIdx.x;
  int stride = gridDim.x * blockDim.x;
  for (; i < n; i += stride) dst[i] = (__bf16)src[i];
}

__global__ void init_h(const float* __restrict__ h0,
                       float* __restrict__ hf,
                       __bf16* __restrict__ hb, int n) {
  int i = blockIdx.x * blockDim.x + threadIdx.x;
  if (i < n) { float v = h0[i]; hf[i] = v; hb[i] = (__bf16)v; }
}

// ---------------- WMMA fragment loaders (global, stride = K) ----------------
// A-matrix 16x32 bf16: lane L(0..15) -> row L,   K in {0..7,16..23}
//                      lane L(16..31)-> row L-16, K in {8..15,24..31}
__device__ __forceinline__ v16bf load_a_frag(const __bf16* __restrict__ base,
                                             int row_stride, int lane) {
  int r  = lane & 15;
  int kc = (lane >> 4) << 3;               // 0 or 8
  const __bf16* p = base + (size_t)r * row_stride + kc;
  v8bf lo = *(const v8bf*)(p);
  v8bf hi = *(const v8bf*)(p + 16);
  return __builtin_shufflevector(lo, hi, 0,1,2,3,4,5,6,7,8,9,10,11,12,13,14,15);
}

// B-matrix 32x16 bf16: lane L -> column N = L%16, K in {kc..kc+7, kc+16..kc+23}, kc = (L/16)*8
__device__ __forceinline__ v16bf load_b_frag(const __bf16* __restrict__ base,
                                             int row_stride, int lane) {
  int c  = lane & 15;
  int kc = (lane >> 4) << 3;               // 0 or 8
  const __bf16* p = base + (size_t)c * row_stride + kc;
  v8bf lo = *(const v8bf*)(p);
  v8bf hi = *(const v8bf*)(p + 16);
  return __builtin_shufflevector(lo, hi, 0,1,2,3,4,5,6,7,8,9,10,11,12,13,14,15);
}

// ---------------- WMMA fragment loaders (LDS, stride = LDS_STRIDE) ----------
__device__ __forceinline__ v16bf lds_frag_a(const __bf16* base, int lane) {
  int r  = lane & 15;
  int kc = (lane >> 4) << 3;
  const __bf16* p = base + r * LDS_STRIDE + kc;
  v8bf lo = *(const v8bf*)(p);             // ds_load_b128
  v8bf hi = *(const v8bf*)(p + 16);
  return __builtin_shufflevector(lo, hi, 0,1,2,3,4,5,6,7,8,9,10,11,12,13,14,15);
}

__device__ __forceinline__ v16bf lds_frag_b(const __bf16* base, int lane) {
  int c  = lane & 15;
  int kc = (lane >> 4) << 3;
  const __bf16* p = base + c * LDS_STRIDE + kc;
  v8bf lo = *(const v8bf*)(p);
  v8bf hi = *(const v8bf*)(p + 16);
  return __builtin_shufflevector(lo, hi, 0,1,2,3,4,5,6,7,8,9,10,11,12,13,14,15);
}

// ---------------- async global -> LDS (CDNA5, tracked by ASYNCcnt) ----------
__device__ __forceinline__ void async_b128(uint32_t lds_byte_off, const void* gaddr) {
  asm volatile("global_load_async_to_lds_b128 %0, %1, off"
               :: "v"(lds_byte_off), "v"(gaddr) : "memory");
}

// Issue one 64x32 A tile + one 128x32 B tile (3 async b128 per thread).
__device__ __forceinline__ void issue_tile(const __bf16* __restrict__ A,
                                           const __bf16* __restrict__ Bm,
                                           int K, int bm, int bn, int k,
                                           uint32_t ldsA, uint32_t ldsB, int tid) {
  {
    int r = tid >> 2, c = (tid & 3) << 3;
    async_b128(ldsA + (uint32_t)(r * LDS_STRIDE + c) * 2,
               (const void*)(A + (size_t)(bm + r) * K + k + c));
  }
#pragma unroll
  for (int rep = 0; rep < 2; ++rep) {
    int idx = tid + (rep << 8);
    int r = idx >> 2, c = (idx & 3) << 3;
    async_b128(ldsB + (uint32_t)(r * LDS_STRIDE + c) * 2,
               (const void*)(Bm + (size_t)(bn + r) * K + k + c));
  }
}

// ---------------- big input-projection GEMM --------------------------------
// C[M,N](bf16) = A[M,K](bf16) @ B[N,K](bf16)^T + bias[N]
// block = 256 thr (8 waves); block tile 64(M) x 128(N); wave tile 32x32.
// Double-buffered async global->LDS pipeline.
__global__ __launch_bounds__(256)
void wmma_gemm_bf16(const __bf16* __restrict__ A,
                    const __bf16* __restrict__ Bm,
                    const float*  __restrict__ bias,
                    __bf16* __restrict__ C,
                    int M, int N, int K) {
  __shared__ __bf16 smem[2][TILE_ROWS * LDS_STRIDE];

  const int tid  = threadIdx.x;
  const int lane = tid & 31;
  const int wave = tid >> 5;
  const int wm = wave & 1;       // 0..1
  const int wn = wave >> 1;      // 0..3
  const int bm = blockIdx.x * 64;
  const int bn = blockIdx.y * 128;
  const int m0l = wm * 32;       // local row inside A tile
  const int n0l = wn * 32;       // local row inside B tile

  const uint32_t lds0 = (uint32_t)(uintptr_t)&smem[0][0];
  const uint32_t ldsA[2] = {lds0, lds0 + (uint32_t)(TILE_ROWS * LDS_STRIDE * 2)};
  const uint32_t ldsBo   = (uint32_t)(64 * LDS_STRIDE * 2);

  v8f acc00 = {}, acc01 = {}, acc10 = {}, acc11 = {};

  const int NK = K >> 5;                  // 32-wide K slabs
  issue_tile(A, Bm, K, bm, bn, 0, ldsA[0], ldsA[0] + ldsBo, tid);

  for (int kk = 0; kk < NK; ++kk) {
    const int buf = kk & 1;
    if (kk + 1 < NK) {
      issue_tile(A, Bm, K, bm, bn, (kk + 1) << 5,
                 ldsA[buf ^ 1], ldsA[buf ^ 1] + ldsBo, tid);
      asm volatile("s_wait_asynccnt 0x3" ::: "memory");   // current tile resident
    } else {
      asm volatile("s_wait_asynccnt 0x0" ::: "memory");
    }
    __syncthreads();

    const __bf16* sA = &smem[buf][0];
    const __bf16* sB = &smem[buf][64 * LDS_STRIDE];
    v16bf a0 = lds_frag_a(sA + m0l * LDS_STRIDE, lane);
    v16bf a1 = lds_frag_a(sA + (m0l + 16) * LDS_STRIDE, lane);
    v16bf b0 = lds_frag_b(sB + n0l * LDS_STRIDE, lane);
    v16bf b1 = lds_frag_b(sB + (n0l + 16) * LDS_STRIDE, lane);
    acc00 = __builtin_amdgcn_wmma_f32_16x16x32_bf16(false, a0, false, b0, (short)0, acc00, false, false);
    acc01 = __builtin_amdgcn_wmma_f32_16x16x32_bf16(false, a0, false, b1, (short)0, acc01, false, false);
    acc10 = __builtin_amdgcn_wmma_f32_16x16x32_bf16(false, a1, false, b0, (short)0, acc10, false, false);
    acc11 = __builtin_amdgcn_wmma_f32_16x16x32_bf16(false, a1, false, b1, (short)0, acc11, false, false);

    __syncthreads();   // everyone done reading buf before it is refilled
  }

  // C/D layout: VGPR v -> M = v + 8*(lane/16), N = lane%16
  const int lh = lane >> 4, lc = lane & 15;
  v8f accs[2][2] = {{acc00, acc01}, {acc10, acc11}};
#pragma unroll
  for (int im = 0; im < 2; ++im) {
#pragma unroll
    for (int in = 0; in < 2; ++in) {
      const int gmb = bm + m0l + im * 16 + lh * 8;
      const int gn  = bn + n0l + in * 16 + lc;
      const float bs = bias[gn];
#pragma unroll
      for (int v = 0; v < 8; ++v)
        C[(size_t)(gmb + v) * N + gn] = (__bf16)(accs[im][in][v] + bs);
    }
  }
}

// ---------------- fused recurrent step -------------------------------------
// Per block: 64 hidden units j. Computes gh_{r,z,n}[:, j..j+63] via WMMA over
// K=H, then gate combine in-lane (r/z/n tiles share (M,N) lane mapping).
__global__ __launch_bounds__(256)
void gru_step(const __bf16* __restrict__ hb_in,   // [B,H] bf16 hidden (read)
              const float*  __restrict__ hf_in,   // [B,H] f32 hidden (read)
              const __bf16* __restrict__ gi,      // [B*T, 3H] bf16 input proj
              const __bf16* __restrict__ whh,     // [3H, H] bf16
              const float*  __restrict__ bhh,     // [3H]
              float*  __restrict__ seq_out,       // [B,T,H]
              float*  __restrict__ hf_out,        // [B,H]
              __bf16* __restrict__ hb_out,        // [B,H]
              int t) {
  const int lane = threadIdx.x & 31;
  const int wave = threadIdx.x >> 5;
  const int mt = wave & 1;                 // batch tile 0..1  (B=32)
  const int nt = wave >> 1;                // j sub-tile 0..3
  const int m0 = mt * 16;
  const int j0 = blockIdx.x * 64 + nt * 16;

  v8f acc_r = {}, acc_z = {}, acc_n = {};

  for (int k = 0; k < H_; k += 32) {
    v16bf a   = load_a_frag(hb_in + (size_t)m0 * H_ + k, H_, lane);
    v16bf br_ = load_b_frag(whh + (size_t)(0 * H_ + j0) * H_ + k, H_, lane);
    v16bf bz_ = load_b_frag(whh + (size_t)(1 * H_ + j0) * H_ + k, H_, lane);
    v16bf bn_ = load_b_frag(whh + (size_t)(2 * H_ + j0) * H_ + k, H_, lane);
    acc_r = __builtin_amdgcn_wmma_f32_16x16x32_bf16(false, a, false, br_, (short)0, acc_r, false, false);
    acc_z = __builtin_amdgcn_wmma_f32_16x16x32_bf16(false, a, false, bz_, (short)0, acc_z, false, false);
    acc_n = __builtin_amdgcn_wmma_f32_16x16x32_bf16(false, a, false, bn_, (short)0, acc_n, false, false);
  }

  const int lh = lane >> 4, lc = lane & 15;
  const int j = j0 + lc;
  const float b_r = bhh[j], b_z = bhh[H_ + j], b_n = bhh[2 * H_ + j];

#pragma unroll
  for (int v = 0; v < 8; ++v) {
    const int b = m0 + lh * 8 + v;                       // batch index
    const size_t girow = ((size_t)b * T_ + t) * G3H;
    const float i_r = (float)gi[girow + j];
    const float i_z = (float)gi[girow + H_ + j];
    const float i_n = (float)gi[girow + 2 * H_ + j];
    const float h_r = acc_r[v] + b_r;
    const float h_z = acc_z[v] + b_z;
    const float h_n = acc_n[v] + b_n;
    const float r = 1.f / (1.f + __expf(-(i_r + h_r)));
    const float z = 1.f / (1.f + __expf(-(i_z + h_z)));
    const float n = tanhf(i_n + r * h_n);
    const float hp = hf_in[(size_t)b * H_ + j];
    const float hnew = (1.f - z) * n + z * hp;
    seq_out[((size_t)b * T_ + t) * H_ + j] = hnew;
    hf_out[(size_t)b * H_ + j] = hnew;
    hb_out[(size_t)b * H_ + j] = (__bf16)hnew;
  }
}

// ---------------- host orchestration ---------------------------------------

extern "C" void kernel_launch(void* const* d_in, const int* in_sizes, int n_in,
                              void* d_out, int out_size, void* d_ws, size_t ws_size,
                              hipStream_t stream) {
  const float* x   = (const float*)d_in[0];   // [B,T,H]
  const float* h0  = (const float*)d_in[1];   // [L,B,H]
  const float* wih = (const float*)d_in[2];   // [L,3H,H]
  const float* whh = (const float*)d_in[3];   // [L,3H,H]
  const float* bih = (const float*)d_in[4];   // [L,3H]
  const float* bhh = (const float*)d_in[5];   // [L,3H]

  float* out_seq = (float*)d_out;                       // [B,T,H]
  float* out_hn  = out_seq + (size_t)B_ * T_ * H_;      // [L,B,H]

  // workspace carve-out
  char* ws = (char*)d_ws;
  size_t off = 0;
  auto take = [&](size_t bytes) -> char* {
    char* p = ws + off;
    off = (off + bytes + 255) & ~(size_t)255;
    return p;
  };
  const size_t WELEMS = (size_t)L_ * G3H * H_;
  __bf16* wih_b = (__bf16*)take(WELEMS * sizeof(__bf16));
  __bf16* whh_b = (__bf16*)take(WELEMS * sizeof(__bf16));
  __bf16* xb    = (__bf16*)take((size_t)B_ * T_ * H_ * sizeof(__bf16));
  __bf16* gib   = (__bf16*)take((size_t)B_ * T_ * G3H * sizeof(__bf16));
  float*  seqf  = (float*) take((size_t)B_ * T_ * H_ * sizeof(float));
  float*  hf0   = (float*) take((size_t)B_ * H_ * sizeof(float));
  float*  hf1   = (float*) take((size_t)B_ * H_ * sizeof(float));
  __bf16* hb0   = (__bf16*)take((size_t)B_ * H_ * sizeof(__bf16));
  __bf16* hb1   = (__bf16*)take((size_t)B_ * H_ * sizeof(__bf16));
  float*  hf[2]  = {hf0, hf1};
  __bf16* hb[2]  = {hb0, hb1};

  // one-time weight conversion f32 -> bf16
  cvt_f32_bf16<<<4096, 256, 0, stream>>>(wih, wih_b, (int)WELEMS);
  cvt_f32_bf16<<<4096, 256, 0, stream>>>(whh, whh_b, (int)WELEMS);

  const float* seq_in = x;
  const int M = B_ * T_;                 // 16384

  for (int l = 0; l < L_; ++l) {
    // activation convert for this layer's input sequence
    cvt_f32_bf16<<<4096, 256, 0, stream>>>(seq_in, xb, M * H_);

    // gi = seq_in @ w_ih^T + b_ih     (bf16 store, f32 accumulate)
    dim3 gg(M / 64, G3H / 128);
    wmma_gemm_bf16<<<gg, 256, 0, stream>>>(xb,
                                           wih_b + (size_t)l * G3H * H_,
                                           bih + (size_t)l * G3H,
                                           gib, M, G3H, H_);

    // hidden-state init for this layer
    init_h<<<(B_ * H_ + 255) / 256, 256, 0, stream>>>(
        h0 + (size_t)l * B_ * H_, hf[0], hb[0], B_ * H_);

    float* layer_out = (l == L_ - 1) ? out_seq : seqf;
    int cur = 0;
    for (int t = 0; t < T_; ++t) {
      gru_step<<<H_ / 64, 256, 0, stream>>>(
          hb[cur], hf[cur], gib,
          whh_b + (size_t)l * G3H * H_,
          bhh + (size_t)l * G3H,
          layer_out, hf[1 - cur], hb[1 - cur], t);
      cur = 1 - cur;
    }

    // final hidden state of this layer -> h_n[l]
    hipMemcpyAsync(out_hn + (size_t)l * B_ * H_, hf[cur],
                   (size_t)B_ * H_ * sizeof(float),
                   hipMemcpyDeviceToDevice, stream);

    seq_in = layer_out;
  }
}
